// Controller_43696997269775
// MI455X (gfx1250) — matrix operand order repaired and checked by reading, not compiled, not run on hardware
//
#include <hip/hip_runtime.h>

#define HID    1024
#define NGATES 4096    // 4*HID
#define KDIM   HID
#define NK     (KDIM / 32)

typedef __attribute__((ext_vector_type(16))) __bf16       v16bf;
typedef __attribute__((ext_vector_type(8)))  float        v8f;
typedef __attribute__((ext_vector_type(4)))  unsigned int uintx4;

union AFrag { uintx4 u[2]; v16bf v; };   // 32 bytes = 16 bf16 = one wave-lane fragment

__device__ __forceinline__ unsigned short f32_to_bf16_rne(float f) {
  unsigned int u = __float_as_uint(f);
  unsigned int r = 0x7FFFu + ((u >> 16) & 1u);   // round-to-nearest-even
  u += r;
  return (unsigned short)(u >> 16);
}

// async global -> LDS copy of 16 bytes (ASYNCcnt-tracked, GV addressing)
__device__ __forceinline__ void async_copy_b128(unsigned lds_byte_off,
                                                const void* gptr) {
  asm volatile("global_load_async_to_lds_b128 %0, %1, off"
               :: "v"(lds_byte_off),
                  "v"((unsigned long long)(uintptr_t)gptr)
               : "memory");
}

// ---------------------------------------------------------------------------
// 1) generic f32 -> bf16 converter, 4 elements / thread
// ---------------------------------------------------------------------------
__global__ void cvt_bf16_kernel(const float* __restrict__ in,
                                unsigned short* __restrict__ out, int n4) {
  int t = blockIdx.x * blockDim.x + threadIdx.x;
  if (t >= n4) return;
  float4 v = ((const float4*)in)[t];
  ushort4 o;
  o.x = f32_to_bf16_rne(v.x);
  o.y = f32_to_bf16_rne(v.y);
  o.z = f32_to_bf16_rne(v.z);
  o.w = f32_to_bf16_rne(v.w);
  ((ushort4*)out)[t] = o;
}

// ---------------------------------------------------------------------------
// 2) embedding gather + hx convert -> bf16   (one block of 256 thr per row)
// ---------------------------------------------------------------------------
__global__ void gather_cvt_kernel(const int* __restrict__ x,
                                  const float* __restrict__ embed,
                                  const float* __restrict__ hx,
                                  unsigned short* __restrict__ Ebf,
                                  unsigned short* __restrict__ Hbf) {
  int t = blockIdx.x * blockDim.x + threadIdx.x;   // B * (HID/4) threads
  int b = t >> 8;            // HID/4 == 256
  int k = (t & 255) << 2;
  int row = x[b];
  float4 ev = *(const float4*)(embed + (size_t)row * HID + k);
  float4 hv = *(const float4*)(hx    + (size_t)b   * HID + k);
  ushort4 eo, ho;
  eo.x = f32_to_bf16_rne(ev.x); eo.y = f32_to_bf16_rne(ev.y);
  eo.z = f32_to_bf16_rne(ev.z); eo.w = f32_to_bf16_rne(ev.w);
  ho.x = f32_to_bf16_rne(hv.x); ho.y = f32_to_bf16_rne(hv.y);
  ho.z = f32_to_bf16_rne(hv.z); ho.w = f32_to_bf16_rne(hv.w);
  *(ushort4*)(Ebf + (size_t)b * HID + k) = eo;
  *(ushort4*)(Hbf + (size_t)b * HID + k) = ho;
}

// ---------------------------------------------------------------------------
// 3) fused dual GEMM on WMMA: gates = E * Wih^T + H * Whh^T
//    block = 8 waves (2M x 4N), block tile 128x256, wave tile 64x64.
//    A/B tiles double-buffered in LDS via global_load_async_to_lds_b128.
// ---------------------------------------------------------------------------
__global__ __launch_bounds__(256) void gates_wmma_kernel(
    const unsigned short* __restrict__ E,
    const unsigned short* __restrict__ Hb,
    const unsigned short* __restrict__ Wi,
    const unsigned short* __restrict__ Wh,
    float* __restrict__ gates) {
  // LDS staging: [buf][E|H][128 rows * 32 K] and [buf][Wi|Wh][256 rows * 32 K]
  __shared__ __attribute__((aligned(16))) unsigned short sA[2][2][128 * 32];
  __shared__ __attribute__((aligned(16))) unsigned short sB[2][2][256 * 32];

  const int t    = threadIdx.x;        // 0..255
  const int lane = t & 31;
  const int w    = t >> 5;             // wave id 0..7
  const int wm   = w >> 2;             // 0..1   (M direction)
  const int wn   = w & 3;              // 0..3   (N direction)
  const int llo  = lane & 15;
  const int lhi  = lane >> 4;

  const size_t M0 = (size_t)blockIdx.y * 128;
  const size_t N0 = (size_t)blockIdx.x * 256;
  const int    aoff = lhi * 8;         // A frag: K = lhi*8..+7 and 16+lhi*8..+7
  const int    boff = lhi * 8;         // B frag: K = lhi*8..+7 and 16+lhi*8..+7

  // issue one 32-K stage of async copies (12 x b128 per thread)
  auto issue_stage = [&](int k0, int bf) {
#pragma unroll
    for (int r = 0; r < 2; ++r) {      // E, H: 128x32 = 512 chunks of 8 bf16
      int c   = t + r * 256;
      int row = c >> 2;
      int col = (c & 3) * 8;
      async_copy_b128((unsigned)(uintptr_t)&sA[bf][0][c * 8],
                      E  + (M0 + row) * KDIM + k0 + col);
      async_copy_b128((unsigned)(uintptr_t)&sA[bf][1][c * 8],
                      Hb + (M0 + row) * KDIM + k0 + col);
    }
#pragma unroll
    for (int r = 0; r < 4; ++r) {      // Wi, Wh: 256x32 = 1024 chunks
      int c   = t + r * 256;
      int row = c >> 2;
      int col = (c & 3) * 8;
      async_copy_b128((unsigned)(uintptr_t)&sB[bf][0][c * 8],
                      Wi + (N0 + row) * KDIM + k0 + col);
      async_copy_b128((unsigned)(uintptr_t)&sB[bf][1][c * 8],
                      Wh + (N0 + row) * KDIM + k0 + col);
    }
  };

  v8f acc[4][4] = {};

  issue_stage(0, 0);                   // prologue: stage 0 -> buffer 0

  for (int k = 0; k < NK; ++k) {
    const int cur = k & 1;
    if (k + 1 < NK) {
      issue_stage((k + 1) * 32, cur ^ 1);            // overlap next stage
      asm volatile("s_wait_asynccnt 0xc" ::: "memory");  // stage k landed
    } else {
      asm volatile("s_wait_asynccnt 0x0" ::: "memory");
    }
    __syncthreads();                   // all waves' stage-k data visible

    AFrag a[4], bfr[4];
    // ---- phase 1: E x Wih^T ----
#pragma unroll
    for (int i = 0; i < 4; ++i) {
      const unsigned short* p = &sA[cur][0][(wm * 64 + i * 16 + llo) * 32 + aoff];
      a[i].u[0] = *(const uintx4*)p;         // ds_load_b128
      a[i].u[1] = *(const uintx4*)(p + 16);
    }
#pragma unroll
    for (int j = 0; j < 4; ++j) {
      const unsigned short* p = &sB[cur][0][(wn * 64 + j * 16 + llo) * 32 + boff];
      bfr[j].u[0] = *(const uintx4*)p;
      bfr[j].u[1] = *(const uintx4*)(p + 16);
    }
#pragma unroll
    for (int i = 0; i < 4; ++i)
#pragma unroll
      for (int j = 0; j < 4; ++j)
        acc[i][j] = __builtin_amdgcn_wmma_f32_16x16x32_bf16(
            false, a[i].v, false, bfr[j].v, (short)0, acc[i][j], false, false);

    // ---- phase 2: H x Whh^T (same accumulators) ----
#pragma unroll
    for (int i = 0; i < 4; ++i) {
      const unsigned short* p = &sA[cur][1][(wm * 64 + i * 16 + llo) * 32 + aoff];
      a[i].u[0] = *(const uintx4*)p;
      a[i].u[1] = *(const uintx4*)(p + 16);
    }
#pragma unroll
    for (int j = 0; j < 4; ++j) {
      const unsigned short* p = &sB[cur][1][(wn * 64 + j * 16 + llo) * 32 + boff];
      bfr[j].u[0] = *(const uintx4*)p;
      bfr[j].u[1] = *(const uintx4*)(p + 16);
    }
#pragma unroll
    for (int i = 0; i < 4; ++i)
#pragma unroll
      for (int j = 0; j < 4; ++j)
        acc[i][j] = __builtin_amdgcn_wmma_f32_16x16x32_bf16(
            false, a[i].v, false, bfr[j].v, (short)0, acc[i][j], false, false);

    __syncthreads();                   // buffer reusable for stage k+2
  }

  // ---- store: C/D layout: VGPR r -> M = r + lhi*8, N = llo ----
  const size_t m0 = M0 + (size_t)wm * 64;
  const size_t n0 = N0 + (size_t)wn * 64;
#pragma unroll
  for (int i = 0; i < 4; ++i)
#pragma unroll
    for (int j = 0; j < 4; ++j) {
      float* gp = gates + (m0 + (size_t)i * 16 + (size_t)lhi * 8) * NGATES
                        + n0 + (size_t)j * 16 + llo;
#pragma unroll
      for (int r = 0; r < 8; ++r)
        gp[(size_t)r * NGATES] = acc[i][j][r];
    }
}

// ---------------------------------------------------------------------------
// 4) pointwise LSTM cell (f32): i,f,o = sigmoid; g = tanh
// ---------------------------------------------------------------------------
__global__ void lstm_pointwise_kernel(const float* __restrict__ gates,
                                      const float* __restrict__ cx,
                                      const float* __restrict__ b_ih,
                                      const float* __restrict__ b_hh,
                                      float* __restrict__ hx_new,
                                      float* __restrict__ cx_new, int total) {
  int t = blockIdx.x * blockDim.x + threadIdx.x;
  if (t >= total) return;
  int b = t >> 10;           // / HID
  int h = t & 1023;
  const float* g = gates + (size_t)b * NGATES;
  float iv = g[h]           + b_ih[h]           + b_hh[h];
  float fv = g[HID + h]     + b_ih[HID + h]     + b_hh[HID + h];
  float gv = g[2 * HID + h] + b_ih[2 * HID + h] + b_hh[2 * HID + h];
  float ov = g[3 * HID + h] + b_ih[3 * HID + h] + b_hh[3 * HID + h];
  float ig = 1.0f / (1.0f + __expf(-iv));
  float fg = 1.0f / (1.0f + __expf(-fv));
  float gg = tanhf(gv);
  float og = 1.0f / (1.0f + __expf(-ov));
  float c  = fg * cx[t] + ig * gg;
  float hn = og * tanhf(c);
  cx_new[t] = c;
  hx_new[t] = hn;
}

// ---------------------------------------------------------------------------
// 5) tiny heads: type_logit[b,0:16], mag_logit[b,0:11] (f32 dot products)
// ---------------------------------------------------------------------------
__global__ void heads_kernel(const float* __restrict__ hx_new,
                             const float* __restrict__ Wt, const float* __restrict__ bt,
                             const float* __restrict__ Wm, const float* __restrict__ bm,
                             float* __restrict__ type_out, float* __restrict__ mag_out,
                             int B) {
  int t = blockIdx.x * blockDim.x + threadIdx.x;
  int b = t >> 5;
  int j = t & 31;
  if (b >= B || j >= 27) return;
  const float4* h4 = (const float4*)(hx_new + (size_t)b * HID);
  const float4* w4;
  float s;
  if (j < 16) { w4 = (const float4*)(Wt + (size_t)j * HID);        s = bt[j]; }
  else        { w4 = (const float4*)(Wm + (size_t)(j - 16) * HID); s = bm[j - 16]; }
#pragma unroll 4
  for (int k = 0; k < HID / 4; ++k) {
    float4 hv = h4[k], wv = w4[k];
    s = fmaf(hv.x, wv.x, s);
    s = fmaf(hv.y, wv.y, s);
    s = fmaf(hv.z, wv.z, s);
    s = fmaf(hv.w, wv.w, s);
  }
  if (j < 16) type_out[(size_t)b * 16 + j] = s;
  else        mag_out[(size_t)b * 11 + (j - 16)] = s;
}

// ---------------------------------------------------------------------------
extern "C" void kernel_launch(void* const* d_in, const int* in_sizes, int n_in,
                              void* d_out, int out_size, void* d_ws, size_t ws_size,
                              hipStream_t stream) {
  (void)n_in; (void)out_size; (void)ws_size;
  const int*   x      = (const int*)  d_in[0];
  const float* hx     = (const float*)d_in[1];
  const float* cx     = (const float*)d_in[2];
  /* d_in[3] = batch_size scalar (unused; derived from hx size) */
  const float* embed  = (const float*)d_in[4];
  const float* W_ih   = (const float*)d_in[5];
  const float* W_hh   = (const float*)d_in[6];
  const float* b_ih   = (const float*)d_in[7];
  const float* b_hh   = (const float*)d_in[8];
  const float* W_type = (const float*)d_in[9];
  const float* b_type = (const float*)d_in[10];
  const float* W_mag  = (const float*)d_in[11];
  const float* b_mag  = (const float*)d_in[12];

  const int B = in_sizes[1] / HID;   // 8192

  // workspace layout (~185 MB)
  unsigned short* Ebf  = (unsigned short*)d_ws;
  unsigned short* Hbf  = Ebf  + (size_t)B * HID;
  unsigned short* Wibf = Hbf  + (size_t)B * HID;
  unsigned short* Whbf = Wibf + (size_t)NGATES * HID;
  float*          gates = (float*)(Whbf + (size_t)NGATES * HID);

  // d_out layout: type | mag | hx_new | cx_new
  float* out      = (float*)d_out;
  float* type_out = out;
  float* mag_out  = out + (size_t)B * 16;
  float* hx_new   = out + (size_t)B * 27;
  float* cx_new   = hx_new + (size_t)B * HID;

  {  // 1) weights -> bf16
    int n4 = NGATES * HID / 4;
    cvt_bf16_kernel<<<(n4 + 255) / 256, 256, 0, stream>>>(W_ih, Wibf, n4);
    cvt_bf16_kernel<<<(n4 + 255) / 256, 256, 0, stream>>>(W_hh, Whbf, n4);
  }
  {  // 2) embedding gather + hx -> bf16
    int threads = B * (HID / 4);
    gather_cvt_kernel<<<threads / 256, 256, 0, stream>>>(x, embed, hx, Ebf, Hbf);
  }
  {  // 3) fused dual WMMA GEMM with async-LDS double buffering
    dim3 grid(NGATES / 256, B / 128);   // 16 x 64 block tiles
    gates_wmma_kernel<<<grid, 256, 0, stream>>>(Ebf, Hbf, Wibf, Whbf, gates);
  }
  {  // 4) LSTM pointwise
    int total = B * HID;
    lstm_pointwise_kernel<<<(total + 255) / 256, 256, 0, stream>>>(
        gates, cx, b_ih, b_hh, hx_new, cx_new, total);
  }
  {  // 5) heads
    int total = B * 32;
    heads_kernel<<<(total + 255) / 256, 256, 0, stream>>>(
        hx_new, W_type, b_type, W_mag, b_mag, type_out, mag_out, B);
  }
}
